// LeNet5XNOR_28887950033177
// MI455X (gfx1250) — 16-bit and 8-bit matrix operand order repaired and checked by reading, compiled, not run on hardware
//
#include <hip/hip_runtime.h>
#include <hip/hip_bf16.h>
#include <stdint.h>

typedef __attribute__((ext_vector_type(16))) _Float16 v16h;
typedef __attribute__((ext_vector_type(8)))  float    v8f;
typedef __attribute__((ext_vector_type(8)))  int      v8i;

#define BATCH 4096
#define EPS_A 1e-4f
#define EPS_B 1e-5f

__global__ void k_zero(float* p, int n) {
    for (int i = threadIdx.x; i < n; i += blockDim.x) p[i] = 0.f;
}

__global__ void k_im2col_off(unsigned short* __restrict__ off) {
    int k = blockIdx.x * blockDim.x + threadIdx.x;
    if (k < 512) {
        unsigned short v = 0;
        if (k < 500) {
            int ic = k / 25, rem = k % 25, r = rem / 5, s = rem % 5;
            v = (unsigned short)(ic * 196 + r * 14 + s);
        }
        off[k] = v;
    }
}

__global__ void k_conv1_stats(const float* __restrict__ x, const float* __restrict__ W1,
                              const float* __restrict__ b1, float* __restrict__ stats1) {
    __shared__ float w[500];
    __shared__ float red[40];
    int tid = threadIdx.x;
    for (int i = tid; i < 500; i += blockDim.x) w[i] = W1[i];
    if (tid < 40) red[tid] = 0.f;
    __syncthreads();

    int idx = blockIdx.x * blockDim.x + tid;
    const int total = BATCH * 28 * 28;
    if (idx < total) {
        int b = idx / 784, p = idx % 784, oy = p / 28, ox = p % 28;
        float patch[25];
        const float* xb = x + b * 1024 + oy * 32 + ox;
#pragma unroll
        for (int r = 0; r < 5; ++r)
#pragma unroll
            for (int s = 0; s < 5; ++s) patch[r * 5 + s] = xb[r * 32 + s];
        for (int c = 0; c < 20; ++c) {
            float acc = b1[c];
#pragma unroll
            for (int k = 0; k < 25; ++k) acc += patch[k] * w[c * 25 + k];
            atomicAdd(&red[c], acc);
            atomicAdd(&red[20 + c], acc * acc);
        }
    }
    __syncthreads();
    if (tid < 40) atomicAdd(&stats1[tid], red[tid]);
}

__global__ void k_bn1_final(const float* stats1, float* bn1) {
    int c = threadIdx.x;
    if (c < 20) {
        const float N = (float)(BATCH * 784);
        float mean = stats1[c] / N;
        float var  = stats1[20 + c] / N - mean * mean;
        bn1[c] = mean;
        bn1[20 + c] = rsqrtf(var + EPS_A);
    }
}

__global__ void k_layer1(const float* __restrict__ x, const float* __restrict__ W1,
                         const float* __restrict__ b1, const float* __restrict__ bn1,
                         float* __restrict__ a1, float* __restrict__ stats2) {
    __shared__ float red[40];
    int tid = threadIdx.x;
    if (tid < 40) red[tid] = 0.f;
    __syncthreads();

    int idx = blockIdx.x * blockDim.x + tid;
    const int total = BATCH * 20 * 196;
    if (idx < total) {
        int b = idx / (20 * 196), r0 = idx % (20 * 196);
        int c = r0 / 196, p = r0 % 196, py = p / 14, px = p % 14;
        float w[25];
#pragma unroll
        for (int k = 0; k < 25; ++k) w[k] = W1[c * 25 + k];
        float patch[36];
        const float* xb = x + b * 1024 + (2 * py) * 32 + 2 * px;
#pragma unroll
        for (int r = 0; r < 6; ++r)
#pragma unroll
            for (int s = 0; s < 6; ++s) patch[r * 6 + s] = xb[r * 32 + s];
        float mean = bn1[c], inv = bn1[20 + c], bias = b1[c];
        float best = -1e30f;
#pragma unroll
        for (int dy = 0; dy < 2; ++dy)
#pragma unroll
            for (int dx = 0; dx < 2; ++dx) {
                float acc = bias;
#pragma unroll
                for (int r = 0; r < 5; ++r)
#pragma unroll
                    for (int s = 0; s < 5; ++s)
                        acc += patch[(r + dy) * 6 + (s + dx)] * w[r * 5 + s];
                float v = fmaxf((acc - mean) * inv, 0.f);
                best = fmaxf(best, v);
            }
        a1[idx] = best;
        atomicAdd(&red[c], best);
        atomicAdd(&red[20 + c], best * best);
    }
    __syncthreads();
    if (tid < 40) atomicAdd(&stats2[tid], red[tid]);
}

__global__ void k_bn2_final(const float* stats2, const float* g2, const float* beta2, float* t2) {
    int c = threadIdx.x;
    if (c < 20) {
        const float N = (float)(BATCH * 196);
        float mean = stats2[c] / N;
        float var  = stats2[20 + c] / N - mean * mean;
        t2[c] = mean - beta2[c] * sqrtf(var + EPS_A) / fmaxf(g2[c], 0.01f);
    }
}

__global__ void k_binarize1(const float* __restrict__ a1, const float* __restrict__ t2,
                            signed char* __restrict__ a2) {
    int idx = blockIdx.x * blockDim.x + threadIdx.x;
    const int total = BATCH * 20 * 196;
    if (idx < total) {
        int c = (idx / 196) % 20;
        a2[idx] = (a1[idx] > t2[c]) ? (signed char)1 : (signed char)-1;
    }
}

__global__ void k_alpha2(const float* __restrict__ W2, float* __restrict__ alpha2) {
    int n = blockIdx.x * blockDim.x + threadIdx.x;
    if (n < 50) {
        float s = 0.f;
        for (int k = 0; k < 500; ++k) s += fabsf(W2[n * 500 + k]);
        alpha2[n] = s / 500.f;
    }
}
__global__ void k_pack_w2(const float* __restrict__ W2, signed char* __restrict__ Bp) {
    int idx = blockIdx.x * blockDim.x + threadIdx.x;
    if (idx < 64 * 512) {
        int n = idx / 512, k = idx % 512;
        signed char v = 0;
        if (n < 50 && k < 500) {
            float w = W2[n * 500 + k];
            v = (w > 0.f) ? 1 : ((w < 0.f) ? -1 : 0);
        }
        Bp[idx] = v;
    }
}

__global__ void __launch_bounds__(128)
k_conv2_wmma(const signed char* __restrict__ a2, const signed char* __restrict__ Bp,
             const unsigned short* __restrict__ offt, const float* __restrict__ alpha2,
             const float* __restrict__ b2, float* __restrict__ c2) {
    __shared__ unsigned int Abuf[64 * 16];
    __shared__ unsigned int offs[256];
    __shared__ int rowbase[64];
    const int tid = threadIdx.x;
    const int wave = tid >> 5, lane = tid & 31;
    const int ln = lane & 15, lh = lane >> 4;
    const int Mbase = blockIdx.x * 64;

    {
        const unsigned int* ot = (const unsigned int*)offt;
        offs[tid] = ot[tid];
        offs[tid + 128] = ot[tid + 128];
    }
    if (tid < 64) {
        int m = Mbase + tid;
        int b = m / 100, p = m % 100, oy = p / 10, ox = p % 10;
        rowbase[tid] = b * 3920 + oy * 14 + ox;
    }
    __syncthreads();

    v8i acc0 = {}, acc1 = {}, acc2 = {}, acc3 = {};
    const signed char* bp = Bp + (wave * 16 + ln) * 512;
#pragma unroll 1
    for (int kt = 0; kt < 8; ++kt) {
        const int kb = kt * 64;
#pragma unroll
        for (int i = 0; i < 8; ++i) {
            int u = tid + 128 * i;
            int row = u >> 4, k0 = kb + (u & 15) * 4;
            unsigned int word = 0;
            if (k0 < 500) {
                int base = rowbase[row];
                unsigned int o01 = offs[k0 >> 1];
                unsigned int o23 = offs[(k0 >> 1) + 1];
                unsigned int b0 = (unsigned char)a2[base + (o01 & 0xffff)];
                unsigned int b1 = (unsigned char)a2[base + (o01 >> 16)];
                unsigned int b2v = (unsigned char)a2[base + (o23 & 0xffff)];
                unsigned int b3 = (unsigned char)a2[base + (o23 >> 16)];
                word = b0 | (b1 << 8) | (b2v << 16) | (b3 << 24);
            }
            Abuf[u] = word;
        }
        __syncthreads();

        v8i B;
#pragma unroll
        for (int v = 0; v < 8; ++v) {
            int koff = (v >> 2) * 32 + lh * 16 + (v & 3) * 4;
            B[v] = *(const int*)(bp + kb + koff);
        }
#pragma unroll
        for (int mt = 0; mt < 4; ++mt) {
            v8i A;
#pragma unroll
            for (int v = 0; v < 8; ++v) {
                int kvb = (v >> 2) * 32 + lh * 16 + (v & 3) * 4;
                A[v] = (int)Abuf[(mt * 16 + ln) * 16 + (kvb >> 2)];
            }
            if (mt == 0) acc0 = __builtin_amdgcn_wmma_i32_16x16x64_iu8(true, A, true, B, acc0, false, false);
            if (mt == 1) acc1 = __builtin_amdgcn_wmma_i32_16x16x64_iu8(true, A, true, B, acc1, false, false);
            if (mt == 2) acc2 = __builtin_amdgcn_wmma_i32_16x16x64_iu8(true, A, true, B, acc2, false, false);
            if (mt == 3) acc3 = __builtin_amdgcn_wmma_i32_16x16x64_iu8(true, A, true, B, acc3, false, false);
        }
        __syncthreads();
    }
    int n = wave * 16 + ln;
    if (n < 50) {
        float aa = alpha2[n], bb = b2[n];
#pragma unroll
        for (int mt = 0; mt < 4; ++mt) {
            v8i acc = (mt == 0) ? acc0 : (mt == 1) ? acc1 : (mt == 2) ? acc2 : acc3;
#pragma unroll
            for (int r = 0; r < 8; ++r) {
                int m = Mbase + mt * 16 + r + lh * 8;
                int b = m / 100, p = m % 100;
                float v = fminf(fmaxf(aa * (float)acc[r] + bb, -1.f), 1.f);
                c2[(b * 50 + n) * 100 + p] = v;
            }
        }
    }
}

__global__ void k_pool2_stats(const float* __restrict__ c2, float* __restrict__ p2,
                              float* __restrict__ stats3) {
    __shared__ float red[100];
    int tid = threadIdx.x;
    if (tid < 100) red[tid] = 0.f;
    __syncthreads();
    int idx = blockIdx.x * blockDim.x + tid;
    const int total = BATCH * 50 * 25;
    if (idx < total) {
        int b = idx / 1250, r0 = idx % 1250, n = r0 / 25, q = r0 % 25;
        int py = q / 5, px = q % 5;
        const float* base = c2 + (b * 50 + n) * 100 + (2 * py) * 10 + 2 * px;
        float v = fmaxf(fmaxf(base[0], base[1]), fmaxf(base[10], base[11]));
        p2[idx] = v;
        atomicAdd(&red[n], v);
        atomicAdd(&red[50 + n], v * v);
    }
    __syncthreads();
    if (tid < 100) atomicAdd(&stats3[tid], red[tid]);
}

__global__ void k_bn3_final(const float* stats3, const float* g3, const float* beta3, float* t3) {
    int c = threadIdx.x;
    if (c < 50) {
        const float N = (float)(BATCH * 25);
        float mean = stats3[c] / N;
        float var  = stats3[50 + c] / N - mean * mean;
        t3[c] = mean - beta3[c] * sqrtf(var + EPS_B) / fmaxf(g3[c], 0.01f);
    }
}

__global__ void k_binarize2(const float* __restrict__ p2, const float* __restrict__ t3,
                            signed char* __restrict__ z) {
    int idx = blockIdx.x * blockDim.x + threadIdx.x;
    const int total = BATCH * 1280;
    if (idx < total) {
        int b = idx / 1280, j = idx % 1280;
        signed char v = 0;
        if (j < 1250) {
            int c = j / 25;
            v = (p2[(b * 50 + c) * 25 + (j % 25)] > t3[c]) ? (signed char)1 : (signed char)-1;
        }
        z[idx] = v;
    }
}

__global__ void k_alphaL(const float* __restrict__ Wl, float* __restrict__ alphaL) {
    int n = blockIdx.x * blockDim.x + threadIdx.x;
    if (n < 500) {
        float s = 0.f;
        for (int k = 0; k < 1250; ++k) s += fabsf(Wl[n * 1250 + k]);
        alphaL[n] = s / 1250.f;
    }
}
__global__ void k_pack_wl(const float* __restrict__ Wl, signed char* __restrict__ Bp) {
    int idx = blockIdx.x * blockDim.x + threadIdx.x;
    if (idx < 512 * 1280) {
        int n = idx / 1280, k = idx % 1280;
        signed char v = 0;
        if (n < 500 && k < 1250) {
            float w = Wl[n * 1250 + k];
            v = (w > 0.f) ? 1 : ((w < 0.f) ? -1 : 0);
        }
        Bp[idx] = v;
    }
}

__global__ void __launch_bounds__(128)
k_linear_wmma(const signed char* __restrict__ z, const signed char* __restrict__ Bp,
              const float* __restrict__ alphaL, const float* __restrict__ bl,
              float* __restrict__ l1) {
    const int tid = threadIdx.x;
    const int wave = tid >> 5, lane = tid & 31;
    const int ln = lane & 15, lh = lane >> 4;
    const int Mbase = blockIdx.x * 64;
    const int Nbase = (blockIdx.y * 4 + wave) * 16;
    const int n = Nbase + ln;
    const signed char* bp = Bp + n * 1280;
    const signed char* za = z + (Mbase + ln) * 1280;

    v8i acc0 = {}, acc1 = {}, acc2 = {}, acc3 = {};
#pragma unroll 1
    for (int kt = 0; kt < 20; ++kt) {
        const int kb = kt * 64;
        v8i B;
#pragma unroll
        for (int v = 0; v < 8; ++v) {
            int koff = (v >> 2) * 32 + lh * 16 + (v & 3) * 4;
            B[v] = *(const int*)(bp + kb + koff);
        }
#pragma unroll
        for (int mt = 0; mt < 4; ++mt) {
            v8i A;
#pragma unroll
            for (int v = 0; v < 8; ++v) {
                int kvb = (v >> 2) * 32 + lh * 16 + (v & 3) * 4;
                A[v] = *(const int*)(za + mt * 16 * 1280 + kb + kvb);
            }
            if (mt == 0) acc0 = __builtin_amdgcn_wmma_i32_16x16x64_iu8(true, A, true, B, acc0, false, false);
            if (mt == 1) acc1 = __builtin_amdgcn_wmma_i32_16x16x64_iu8(true, A, true, B, acc1, false, false);
            if (mt == 2) acc2 = __builtin_amdgcn_wmma_i32_16x16x64_iu8(true, A, true, B, acc2, false, false);
            if (mt == 3) acc3 = __builtin_amdgcn_wmma_i32_16x16x64_iu8(true, A, true, B, acc3, false, false);
        }
    }
    if (n < 500) {
        float aa = alphaL[n], bb = bl[n];
#pragma unroll
        for (int mt = 0; mt < 4; ++mt) {
            v8i acc = (mt == 0) ? acc0 : (mt == 1) ? acc1 : (mt == 2) ? acc2 : acc3;
#pragma unroll
            for (int r = 0; r < 8; ++r) {
                int mr = Mbase + mt * 16 + r + lh * 8;
                l1[mr * 500 + n] = fminf(fmaxf(aa * (float)acc[r] + bb, -1.f), 1.f);
            }
        }
    }
}

__global__ void k_pack_wfc(const float* __restrict__ Wfc, _Float16* __restrict__ Wp) {
    int idx = blockIdx.x * blockDim.x + threadIdx.x;
    if (idx < 16 * 512) {
        int n = idx / 512, k = idx % 512;
        Wp[idx] = (n < 10 && k < 500) ? (_Float16)Wfc[n * 500 + k] : (_Float16)0.f;
    }
}

__global__ void __launch_bounds__(32)
k_fc_wmma(const float* __restrict__ l1, const _Float16* __restrict__ Wp,
          const float* __restrict__ bfc, float* __restrict__ out) {
    const int lane = threadIdx.x;
    const int ln = lane & 15, lh = lane >> 4;
    const int Mbase = blockIdx.x * 16;
    const int m = Mbase + ln;

    v8f acc = {};
#pragma unroll 1
    for (int kt = 0; kt < 16; ++kt) {
        const int kb = kt * 32;
        v16h A, B;
#pragma unroll
        for (int v = 0; v < 8; ++v) {
            int ka = kb + (v >> 2) * 16 + (v & 3) * 2 + lh * 8;
            float f0 = (ka     < 500) ? l1[m * 500 + ka]     : 0.f;
            float f1 = (ka + 1 < 500) ? l1[m * 500 + ka + 1] : 0.f;
            A[2 * v]     = (_Float16)f0;
            A[2 * v + 1] = (_Float16)f1;
            int kbv = kb + (v >> 2) * 16 + (v & 3) * 2 + lh * 8;
            B[2 * v]     = Wp[ln * 512 + kbv];
            B[2 * v + 1] = Wp[ln * 512 + kbv + 1];
        }
        acc = __builtin_amdgcn_wmma_f32_16x16x32_f16(false, A, false, B, (short)0, acc,
                                                     false, false);
    }
    if (ln < 10) {
        float bb = bfc[ln];
#pragma unroll
        for (int r = 0; r < 8; ++r) {
            int mr = Mbase + r + lh * 8;
            out[mr * 10 + ln] = acc[r] + bb;
        }
    }
}

extern "C" void kernel_launch(void* const* d_in, const int* in_sizes, int n_in,
                              void* d_out, int out_size, void* d_ws, size_t ws_size,
                              hipStream_t stream) {
    const float* x    = (const float*)d_in[0];
    const float* W1   = (const float*)d_in[1];
    const float* b1   = (const float*)d_in[2];
    const float* g2   = (const float*)d_in[3];
    const float* be2  = (const float*)d_in[4];
    const float* W2   = (const float*)d_in[5];
    const float* b2   = (const float*)d_in[6];
    const float* g3   = (const float*)d_in[7];
    const float* be3  = (const float*)d_in[8];
    const float* Wl   = (const float*)d_in[9];
    const float* bl   = (const float*)d_in[10];
    const float* Wfc  = (const float*)d_in[11];
    const float* bfc  = (const float*)d_in[12];
    float* out = (float*)d_out;
    char*  ws  = (char*)d_ws;

    size_t o = 0;
    auto alloc = [&](size_t bytes) { size_t r = o; o = (o + bytes + 255) & ~(size_t)255; return r; };
    size_t o_stats1 = alloc(40 * 4);
    size_t o_bn1    = alloc(40 * 4);
    size_t o_stats2 = alloc(40 * 4);
    size_t o_t2     = alloc(20 * 4);
    size_t o_stats3 = alloc(100 * 4);
    size_t o_t3     = alloc(50 * 4);
    size_t o_alpha2 = alloc(50 * 4);
    size_t o_alphaL = alloc(500 * 4);
    size_t o_offt   = alloc(512 * 2);
    size_t o_Bp2    = alloc(64 * 512);
    size_t o_BpL    = alloc(512 * 1280);
    size_t o_Wp     = alloc(16 * 512 * 2);
    size_t o_z      = alloc((size_t)BATCH * 1280);
    size_t o_l1     = alloc((size_t)BATCH * 500 * 4);
    size_t o_regB   = alloc(20480000);
    size_t o_regA   = alloc(81920000);

    float* stats1 = (float*)(ws + o_stats1);
    float* bn1    = (float*)(ws + o_bn1);
    float* stats2 = (float*)(ws + o_stats2);
    float* t2     = (float*)(ws + o_t2);
    float* stats3 = (float*)(ws + o_stats3);
    float* t3     = (float*)(ws + o_t3);
    float* alpha2 = (float*)(ws + o_alpha2);
    float* alphaL = (float*)(ws + o_alphaL);
    unsigned short* offt = (unsigned short*)(ws + o_offt);
    signed char* Bp2 = (signed char*)(ws + o_Bp2);
    signed char* BpL = (signed char*)(ws + o_BpL);
    _Float16* Wp  = (_Float16*)(ws + o_Wp);
    signed char* z   = (signed char*)(ws + o_z);
    float* l1     = (float*)(ws + o_l1);
    signed char* a2  = (signed char*)(ws + o_regB);
    float* p2     = (float*)(ws + o_regB);
    float* a1     = (float*)(ws + o_regA);
    float* c2     = (float*)(ws + o_regA);

    k_zero<<<1, 256, 0, stream>>>(stats1, 40);
    k_zero<<<1, 256, 0, stream>>>(stats2, 40);
    k_zero<<<1, 256, 0, stream>>>(stats3, 100);
    k_im2col_off<<<2, 256, 0, stream>>>(offt);

    k_conv1_stats<<<(BATCH * 784 + 255) / 256, 256, 0, stream>>>(x, W1, b1, stats1);
    k_bn1_final<<<1, 32, 0, stream>>>(stats1, bn1);
    k_layer1<<<(BATCH * 20 * 196 + 255) / 256, 256, 0, stream>>>(x, W1, b1, bn1, a1, stats2);
    k_bn2_final<<<1, 32, 0, stream>>>(stats2, g2, be2, t2);
    k_binarize1<<<(BATCH * 20 * 196 + 255) / 256, 256, 0, stream>>>(a1, t2, a2);

    k_alpha2<<<1, 64, 0, stream>>>(W2, alpha2);
    k_pack_w2<<<(64 * 512 + 255) / 256, 256, 0, stream>>>(W2, Bp2);
    k_conv2_wmma<<<(BATCH * 100) / 64, 128, 0, stream>>>(a2, Bp2, offt, alpha2, b2, c2);
    k_pool2_stats<<<(BATCH * 50 * 25 + 255) / 256, 256, 0, stream>>>(c2, p2, stats3);
    k_bn3_final<<<1, 64, 0, stream>>>(stats3, g3, be3, t3);
    k_binarize2<<<(BATCH * 1280 + 255) / 256, 256, 0, stream>>>(p2, t3, z);

    k_alphaL<<<(500 + 255) / 256, 256, 0, stream>>>(Wl, alphaL);
    k_pack_wl<<<(512 * 1280 + 255) / 256, 256, 0, stream>>>(Wl, BpL);
    dim3 lg(BATCH / 64, 8);
    k_linear_wmma<<<lg, 128, 0, stream>>>(z, BpL, alphaL, bl, l1);

    k_pack_wfc<<<(16 * 512 + 255) / 256, 256, 0, stream>>>(Wfc, Wp);
    k_fc_wmma<<<BATCH / 16, 32, 0, stream>>>(l1, Wp, bfc, out);

    (void)in_sizes; (void)n_in; (void)out_size; (void)ws_size;
}
